// DilateBlock_28226525069832
// MI455X (gfx1250) — compile-verified, matrix operand order checked by reading, not run on hardware
//
#include <hip/hip_runtime.h>
#include <hip/hip_bf16.h>
#include <math.h>

typedef _Float16 h16;
typedef __attribute__((ext_vector_type(16))) _Float16 v16h;
typedef __attribute__((ext_vector_type(8)))  float    v8f;

#define NVOX 13824   // 24^3
#define DIMC 96
#define QKVD 288
#define HIDC 384
#define DSZ  24
#define EPSV 1e-5f

// ---------------------------------------------------------------------------
// Weight pre-pack: fp32 (K x N) row-major -> f16 WMMA B-fragment tiles.
// Tile (kt, ct) holds 32 lanes x 16 halves contiguous; lane = half*16 + col,
// element i => K = kt*32 + half*16 + i, N = ct*16 + col  (contiguous k order, same as the A reads).
// ---------------------------------------------------------------------------
__global__ void pack_b_kernel(const float* __restrict__ W, h16* __restrict__ Bp,
                              int K, int N) {
  const int i = blockIdx.x * blockDim.x + threadIdx.x;
  if (i >= K * N) return;
  const int k = i / N, n = i % N;
  const int kt = k >> 5, half = (k >> 4) & 1, ki = k & 15;
  const int ct = n >> 4, col = n & 15;
  const int lane = half * 16 + col;
  const size_t idx = ((((size_t)kt * (N >> 4) + ct) * 32) + lane) * 16 + ki;
  Bp[idx] = (h16)W[i];
}

// ---------------------------------------------------------------------------
// One wave32 computes a 16x48 strip: 3 WMMAs per A fragment, K fully unrolled.
// A: row-major f16 with lda == K (contiguous rows). Bp: packed tiles (above).
// ---------------------------------------------------------------------------
template<int K, int NCOLS>
__device__ inline void wmma_strip3(const h16* __restrict__ A, const h16* __restrict__ Bp,
                                   int row0, int ct0, int lane, v8f* acc) {
  const int r = row0 + (lane & 15);
  const int half = lane >> 4;
  #pragma unroll
  for (int k0 = 0; k0 < K; k0 += 32) {
    const h16* ap = A + (size_t)r * K + k0 + half * 16;
    v16h a = *(const v16h*)ap;                       // 32B contiguous per lane
    if (k0 + 32 < K) __builtin_prefetch(ap + 32, 0, 0);  // global_prefetch_b8
    #pragma unroll
    for (int j = 0; j < 3; ++j) {
      const h16* bp = Bp + ((((size_t)(k0 >> 5) * (NCOLS >> 4)) + ct0 + j) << 9)
                         + (lane << 4);              // 32B contiguous per lane
      v16h b = *(const v16h*)bp;
      acc[j] = __builtin_amdgcn_wmma_f32_16x16x32_f16(
          false, a, false, b, (short)0, acc[j], false, false);
    }
  }
}

// ---------------------------------------------------------------------------
// LayerNorm over channels-first input -> f16 channels-last rows
__global__ void ln1_kernel(const float* __restrict__ mov, const float* __restrict__ fix,
                           const float* __restrict__ mw, const float* __restrict__ mb,
                           const float* __restrict__ fw, const float* __restrict__ fb,
                           h16* __restrict__ a16) {
  const int s = blockIdx.y;
  const int n = blockIdx.x * blockDim.x + threadIdx.x;
  if (n >= NVOX) return;
  const float* x = s ? fix : mov;
  const float* w = s ? fw : mw;
  const float* b = s ? fb : mb;
  float sum = 0.f, sq = 0.f;
  for (int c = 0; c < DIMC; ++c) { float v = x[(size_t)c * NVOX + n]; sum += v; sq += v * v; }
  const float mu = sum * (1.f / DIMC);
  const float rs = rsqrtf(sq * (1.f / DIMC) - mu * mu + EPSV);
  h16* out = a16 + ((size_t)s * NVOX + n) * DIMC;
  for (int c = 0; c < DIMC; ++c)
    out[c] = (h16)((x[(size_t)c * NVOX + n] - mu) * rs * w[c] + b[c]);
}

// QKV projection: (N x 96) x (96 x 288) + bias -> f16 qkv rows
__global__ void qkv_gemm_kernel(const h16* __restrict__ a16,
                                const h16* __restrict__ bm16, const h16* __restrict__ bf16,
                                const float* __restrict__ bbm, const float* __restrict__ bbf,
                                h16* __restrict__ qkv16) {
  const int s = blockIdx.y;
  const int wid  = blockIdx.x * (blockDim.x >> 5) + (threadIdx.x >> 5);
  const int lane = threadIdx.x & 31;
  const int ns = QKVD / 48;
  if (wid >= (NVOX / 16) * ns) return;
  const int row0 = (wid / ns) * 16, ct0 = (wid % ns) * 3;
  v8f acc[3] = {};
  wmma_strip3<DIMC, QKVD>(a16 + (size_t)s * NVOX * DIMC, s ? bf16 : bm16,
                          row0, ct0, lane, acc);
  const float* bias = s ? bbf : bbm;
  h16* C = qkv16 + (size_t)s * NVOX * QKVD;
  const int half = lane >> 4;
  #pragma unroll
  for (int j = 0; j < 3; ++j) {
    const int cc = (ct0 + j) * 16 + (lane & 15);
    const float bv = bias[cc];
    #pragma unroll
    for (int v = 0; v < 8; ++v)
      C[(size_t)(row0 + v + 8 * half) * QKVD + cc] = (h16)(acc[j][v] + bv);
  }
}

// Dilated 3x3x3 local cross-attention: one thread per (voxel, head).
// Zero-pad semantics: OOB neighbors keep score 0 in softmax, contribute v = 0.
__global__ void attn_kernel(const h16* __restrict__ qkv16, h16* __restrict__ att16) {
  const int so = blockIdx.y;
  const int t = blockIdx.x * blockDim.x + threadIdx.x;
  if (t >= NVOX * 6) return;
  const int n = t % NVOX, g = t / NVOX;
  const int di = g >> 1, dil = di + 1;
  const int cb = di * 32 + (g & 1) * 16;
  const int x = n % DSZ, y = (n / DSZ) % DSZ, z = n / (DSZ * DSZ);
  const h16* Q  = qkv16 + (size_t)(so ^ 1) * NVOX * QKVD;  // cross: q from other stream
  const h16* KV = qkv16 + (size_t)so * NVOX * QKVD;

  const v16h qv = *(const v16h*)(Q + (size_t)n * QKVD + cb);
  float q[16];
  #pragma unroll
  for (int c = 0; c < 16; ++c) q[c] = (float)qv[c];

  float sc[27]; int nbv[27];
  for (int a = 0; a < 3; ++a)
    for (int b = 0; b < 3; ++b)
      for (int l = 0; l < 3; ++l) {
        const int idx = (a * 3 + b) * 3 + l;
        const int zz = z + (a - 1) * dil, yy = y + (b - 1) * dil, xx = x + (l - 1) * dil;
        const bool ok = (unsigned)zz < (unsigned)DSZ && (unsigned)yy < (unsigned)DSZ &&
                        (unsigned)xx < (unsigned)DSZ;
        const int nb = ok ? ((zz * DSZ + yy) * DSZ + xx) : -1;
        nbv[idx] = nb;
        float dot = 0.f;
        if (ok) {
          const v16h kv = *(const v16h*)(KV + (size_t)nb * QKVD + DIMC + cb);
          #pragma unroll
          for (int c = 0; c < 16; ++c) dot += q[c] * (float)kv[c];
        }
        sc[idx] = dot * 0.25f;                 // HD^-0.5
      }
  float mx = sc[0];
  for (int i = 1; i < 27; ++i) mx = fmaxf(mx, sc[i]);
  float den = 0.f;
  for (int i = 0; i < 27; ++i) { sc[i] = __expf(sc[i] - mx); den += sc[i]; }
  const float inv = 1.f / den;
  float acc[16];
  #pragma unroll
  for (int c = 0; c < 16; ++c) acc[c] = 0.f;
  for (int i = 0; i < 27; ++i) {
    if (nbv[i] < 0) continue;
    const v16h vv = *(const v16h*)(KV + (size_t)nbv[i] * QKVD + 2 * DIMC + cb);
    const float wgt = sc[i];
    #pragma unroll
    for (int c = 0; c < 16; ++c) acc[c] += wgt * (float)vv[c];
  }
  h16* op = att16 + ((size_t)so * NVOX + n) * DIMC + cb;
  #pragma unroll
  for (int c = 0; c < 16; ++c) op[c] = (h16)(acc[c] * inv);
}

// Output projection + bias + residual (residual from channels-first input)
__global__ void proj_gemm_kernel(const h16* __restrict__ att16,
                                 const h16* __restrict__ bm16, const h16* __restrict__ bf16,
                                 const float* __restrict__ bbm, const float* __restrict__ bbf,
                                 const float* __restrict__ mov, const float* __restrict__ fix,
                                 float* __restrict__ ybuf) {
  const int s = blockIdx.y;
  const int wid  = blockIdx.x * (blockDim.x >> 5) + (threadIdx.x >> 5);
  const int lane = threadIdx.x & 31;
  const int ns = DIMC / 48;
  if (wid >= (NVOX / 16) * ns) return;
  const int row0 = (wid / ns) * 16, ct0 = (wid % ns) * 3;
  v8f acc[3] = {};
  wmma_strip3<DIMC, DIMC>(att16 + (size_t)s * NVOX * DIMC, s ? bf16 : bm16,
                          row0, ct0, lane, acc);
  const float* bias = s ? bbf : bbm;
  const float* xin = s ? fix : mov;
  float* Y = ybuf + (size_t)s * NVOX * DIMC;
  const int half = lane >> 4;
  #pragma unroll
  for (int j = 0; j < 3; ++j) {
    const int cc = (ct0 + j) * 16 + (lane & 15);
    const float bv = bias[cc];
    #pragma unroll
    for (int v = 0; v < 8; ++v) {
      const int r = row0 + v + 8 * half;
      Y[(size_t)r * DIMC + cc] = acc[j][v] + bv + xin[(size_t)cc * NVOX + r];
    }
  }
}

// LayerNorm over contiguous f32 rows -> f16 rows (shared n2 weights)
__global__ void ln2_kernel(const float* __restrict__ ybuf,
                           const float* __restrict__ w, const float* __restrict__ b,
                           h16* __restrict__ a16) {
  const int s = blockIdx.y;
  const int n = blockIdx.x * blockDim.x + threadIdx.x;
  if (n >= NVOX) return;
  const float* row = ybuf + ((size_t)s * NVOX + n) * DIMC;
  float sum = 0.f, sq = 0.f;
  for (int c = 0; c < DIMC; ++c) { float v = row[c]; sum += v; sq += v * v; }
  const float mu = sum * (1.f / DIMC);
  const float rs = rsqrtf(sq * (1.f / DIMC) - mu * mu + EPSV);
  h16* out = a16 + ((size_t)s * NVOX + n) * DIMC;
  for (int c = 0; c < DIMC; ++c) out[c] = (h16)((row[c] - mu) * rs * w[c] + b[c]);
}

// MLP fc1 + exact GELU -> f16 hidden
__global__ void mlp1_gemm_kernel(const h16* __restrict__ a16, const h16* __restrict__ w1p,
                                 const float* __restrict__ b1, h16* __restrict__ hbuf) {
  const int s = blockIdx.y;
  const int wid  = blockIdx.x * (blockDim.x >> 5) + (threadIdx.x >> 5);
  const int lane = threadIdx.x & 31;
  const int ns = HIDC / 48;
  if (wid >= (NVOX / 16) * ns) return;
  const int row0 = (wid / ns) * 16, ct0 = (wid % ns) * 3;
  v8f acc[3] = {};
  wmma_strip3<DIMC, HIDC>(a16 + (size_t)s * NVOX * DIMC, w1p, row0, ct0, lane, acc);
  h16* H = hbuf + (size_t)s * NVOX * HIDC;
  const int half = lane >> 4;
  #pragma unroll
  for (int j = 0; j < 3; ++j) {
    const int cc = (ct0 + j) * 16 + (lane & 15);
    const float bv = b1[cc];
    #pragma unroll
    for (int v = 0; v < 8; ++v) {
      const float xg = acc[j][v] + bv;
      const float gl = xg * 0.5f * (1.f + erff(xg * 0.70710678118654752f));
      H[(size_t)(row0 + v + 8 * half) * HIDC + cc] = (h16)gl;
    }
  }
}

// MLP fc2 + bias + residual(Y) -> channels-first output
__global__ void mlp2_gemm_kernel(const h16* __restrict__ hbuf, const h16* __restrict__ w2p,
                                 const float* __restrict__ b2, const float* __restrict__ ybuf,
                                 float* __restrict__ out) {
  const int s = blockIdx.y;
  const int wid  = blockIdx.x * (blockDim.x >> 5) + (threadIdx.x >> 5);
  const int lane = threadIdx.x & 31;
  const int ns = DIMC / 48;
  if (wid >= (NVOX / 16) * ns) return;
  const int row0 = (wid / ns) * 16, ct0 = (wid % ns) * 3;
  v8f acc[3] = {};
  wmma_strip3<HIDC, DIMC>(hbuf + (size_t)s * NVOX * HIDC, w2p, row0, ct0, lane, acc);
  const float* Y = ybuf + (size_t)s * NVOX * DIMC;
  const int half = lane >> 4;
  #pragma unroll
  for (int j = 0; j < 3; ++j) {
    const int cc = (ct0 + j) * 16 + (lane & 15);
    const float bv = b2[cc];
    #pragma unroll
    for (int v = 0; v < 8; ++v) {
      const int r = row0 + v + 8 * half;
      out[(size_t)s * DIMC * NVOX + (size_t)cc * NVOX + r] =
          Y[(size_t)r * DIMC + cc] + acc[j][v] + bv;
    }
  }
}

// ---------------------------------------------------------------------------
extern "C" void kernel_launch(void* const* d_in, const int* in_sizes, int n_in,
                              void* d_out, int out_size, void* d_ws, size_t ws_size,
                              hipStream_t stream) {
  const float* mov   = (const float*)d_in[0];
  const float* fix   = (const float*)d_in[1];
  const float* mn1_w = (const float*)d_in[2];
  const float* mn1_b = (const float*)d_in[3];
  const float* fn1_w = (const float*)d_in[4];
  const float* fn1_b = (const float*)d_in[5];
  const float* Wmqkv = (const float*)d_in[6];
  const float* bmqkv = (const float*)d_in[7];
  const float* Wfqkv = (const float*)d_in[8];
  const float* bfqkv = (const float*)d_in[9];
  const float* Wmp   = (const float*)d_in[10];
  const float* bmp   = (const float*)d_in[11];
  const float* Wfp   = (const float*)d_in[12];
  const float* bfp   = (const float*)d_in[13];
  const float* n2_w  = (const float*)d_in[14];
  const float* n2_b  = (const float*)d_in[15];
  const float* W1    = (const float*)d_in[16];
  const float* b1    = (const float*)d_in[17];
  const float* W2    = (const float*)d_in[18];
  const float* b2    = (const float*)d_in[19];
  float* out = (float*)d_out;

  char* ws = (char*)d_ws;
  size_t off = 0;
  auto carve = [&](size_t bytes) { char* p = ws + off; off = (off + bytes + 255) & ~(size_t)255; return p; };
  h16* wqkv_m16 = (h16*)carve((size_t)DIMC * QKVD * 2);
  h16* wqkv_f16 = (h16*)carve((size_t)DIMC * QKVD * 2);
  h16* wp_m16   = (h16*)carve((size_t)DIMC * DIMC * 2);
  h16* wp_f16   = (h16*)carve((size_t)DIMC * DIMC * 2);
  h16* w1_16    = (h16*)carve((size_t)DIMC * HIDC * 2);
  h16* w2_16    = (h16*)carve((size_t)HIDC * DIMC * 2);
  h16* a16      = (h16*)carve((size_t)2 * NVOX * DIMC * 2);
  h16* qkv16    = (h16*)carve((size_t)2 * NVOX * QKVD * 2);
  h16* att16    = (h16*)carve((size_t)2 * NVOX * DIMC * 2);
  float* ybuf   = (float*)carve((size_t)2 * NVOX * DIMC * 4);
  h16* hbuf     = (h16*)carve((size_t)2 * NVOX * HIDC * 2);

  const int TB = 256;
  auto pack = [&](const float* s, h16* d, int K, int N) {
    pack_b_kernel<<<(K * N + TB - 1) / TB, TB, 0, stream>>>(s, d, K, N);
  };
  pack(Wmqkv, wqkv_m16, DIMC, QKVD);
  pack(Wfqkv, wqkv_f16, DIMC, QKVD);
  pack(Wmp,   wp_m16,   DIMC, DIMC);
  pack(Wfp,   wp_f16,   DIMC, DIMC);
  pack(W1,    w1_16,    DIMC, HIDC);
  pack(W2,    w2_16,    HIDC, DIMC);

  dim3 lng(NVOX / TB, 2);
  ln1_kernel<<<lng, TB, 0, stream>>>(mov, fix, mn1_w, mn1_b, fn1_w, fn1_b, a16);

  const int WPB = TB / 32;   // 8 waves per block, one 16x48 strip per wave
  auto strips = [&](int ncols) { return (NVOX / 16) * (ncols / 48); };

  qkv_gemm_kernel<<<dim3((strips(QKVD) + WPB - 1) / WPB, 2), TB, 0, stream>>>(
      a16, wqkv_m16, wqkv_f16, bmqkv, bfqkv, qkv16);

  attn_kernel<<<dim3((NVOX * 6 + TB - 1) / TB, 2), TB, 0, stream>>>(qkv16, att16);

  proj_gemm_kernel<<<dim3((strips(DIMC) + WPB - 1) / WPB, 2), TB, 0, stream>>>(
      att16, wp_m16, wp_f16, bmp, bfp, mov, fix, ybuf);

  ln2_kernel<<<lng, TB, 0, stream>>>(ybuf, n2_w, n2_b, a16);

  mlp1_gemm_kernel<<<dim3((strips(HIDC) + WPB - 1) / WPB, 2), TB, 0, stream>>>(
      a16, w1_16, b1, hbuf);

  mlp2_gemm_kernel<<<dim3((strips(DIMC) + WPB - 1) / WPB, 2), TB, 0, stream>>>(
      hbuf, w2_16, b2, ybuf, out);
}
